// PhiAttention_22230750724276
// MI455X (gfx1250) — matrix operand order repaired and checked by reading, not compiled, not run on hardware
//
#include <hip/hip_runtime.h>
#include <stdint.h>
#include <stddef.h>

// ---- problem constants (from reference) ----
#define B_   2
#define S_   2048
#define D_   2048
#define H_   32
#define HKV_ 8
#define HD_  64
#define G_   (H_ / HKV_)

typedef __bf16 bf16_t;
typedef __attribute__((ext_vector_type(16))) __bf16  v16bf;
typedef __attribute__((ext_vector_type(8)))  float   v8f;
typedef __attribute__((ext_vector_type(4)))  unsigned int u32x4;
typedef __attribute__((ext_vector_type(2)))  unsigned int u32x2;
typedef __attribute__((ext_vector_type(4)))  float   f32x4;

// 32-byte fragment: 16 bf16 operand for v_wmma_f32_16x16x32_bf16
union Frag16 {
    v16bf v;
    u32x4 u[2];
};
static_assert(sizeof(Frag16) == 32, "frag size");

__device__ __forceinline__ v8f wmma_bf16(const v16bf& a, const v16bf& b, const v8f& c) {
    return __builtin_amdgcn_wmma_f32_16x16x32_bf16(false, a, false, b, (short)0, c,
                                                   false, false);
}

// ---------------------------------------------------------------------------
// fp32 -> bf16 bulk convert (x and all weights, done once)
// ---------------------------------------------------------------------------
__global__ void cvt_f32_bf16(const float* __restrict__ in, bf16_t* __restrict__ out, int n4) {
    int i = blockIdx.x * blockDim.x + threadIdx.x;
    if (i < n4) {
        f32x4 f = ((const f32x4*)in)[i];
        union { bf16_t h[4]; u32x2 u; } t;
        t.h[0] = (bf16_t)f.x; t.h[1] = (bf16_t)f.y;
        t.h[2] = (bf16_t)f.z; t.h[3] = (bf16_t)f.w;
        ((u32x2*)out)[i] = t.u;
    }
}

// ---------------------------------------------------------------------------
// GEMM: out[m,n] = sum_k A[m,k]*W[n,k] + bias[n]      (A, W both bf16)
//   mode 0: write bf16 head-major (b, head, s, hd)   [for Q/K/V]
//   mode 1: write fp32 row-major  (m, n)             [for O-proj -> d_out]
// block = 256 threads = 8 waves arranged 2(M) x 4(N); block tile 64(M)x128(N);
// wave tile 32x32 -> 4 v_wmma per K-step of 32.  LDS double-buffered so the
// next tile's global loads overlap the current tile's WMMAs; one barrier/step.
// ---------------------------------------------------------------------------
__global__ __launch_bounds__(256)
void gemm_wmma(const bf16_t* __restrict__ A, const bf16_t* __restrict__ W,
               const float* __restrict__ bias, bf16_t* __restrict__ outT,
               float* __restrict__ outF, int M, int N, int K, int mode, int nh)
{
    __shared__ bf16_t As[2][64][32];
    __shared__ bf16_t Ws[2][128][32];

    const int tid  = threadIdx.x;
    const int w    = tid >> 5;
    const int L    = tid & 31;
    const int l15  = L & 15;
    const int hi   = L >> 4;         // 0: lanes 0-15, 1: lanes 16-31
    const int wrow = w >> 2;         // 0..1 -> M sub-tile (32 rows)
    const int wcol = w & 3;          // 0..3 -> N sub-tile (32 cols)
    const int mb   = blockIdx.y * 64;
    const int nb   = blockIdx.x * 128;

    const v8f vzero = {0,0,0,0,0,0,0,0};
    v8f acc[2][2];
    acc[0][0] = vzero; acc[0][1] = vzero; acc[1][0] = vzero; acc[1][1] = vzero;

    // staging assignments
    const int ar = tid >> 2;            // 0..63   A row
    const int ac = (tid & 3) * 8;       // 0,8,16,24
    const int wr_ = tid >> 1;           // 0..127  W row
    const int wc_ = (tid & 1) * 16;     // 0,16

    const bf16_t* Aptr = &A[(size_t)(mb + ar) * K + ac];
    const bf16_t* Wptr = &W[(size_t)(nb + wr_) * K + wc_];

    const int nk = K / 32;
    u32x4 a_reg, w_reg0, w_reg1;

    // prologue: stage k-tile 0 into buffer 0
    a_reg  = *(const u32x4*)(Aptr);
    w_reg0 = *(const u32x4*)(Wptr);
    w_reg1 = *(const u32x4*)(Wptr + 8);
    *(u32x4*)&As[0][ar][ac]       = a_reg;
    *(u32x4*)&Ws[0][wr_][wc_]     = w_reg0;
    *(u32x4*)&Ws[0][wr_][wc_ + 8] = w_reg1;
    __syncthreads();

    for (int kt = 0; kt < nk; ++kt) {
        const int cur = kt & 1;
        const bool have_next = (kt + 1) < nk;

        if (have_next) {
            const int kb = (kt + 1) * 32;
            a_reg  = *(const u32x4*)(Aptr + kb);
            w_reg0 = *(const u32x4*)(Wptr + kb);
            w_reg1 = *(const u32x4*)(Wptr + kb + 8);
            if ((kt + 2) < nk) {   // gfx1250 global_prefetch for k-tile after next
                __builtin_prefetch(Aptr + kb + 32, 0, 1);
                __builtin_prefetch(Wptr + kb + 32, 0, 1);
            }
        }

        // ---- compute from buffer `cur` ----
        Frag16 af[2], bf[2];
        #pragma unroll
        for (int s = 0; s < 2; ++s) {
            af[s].u[0] = *(const u32x4*)&As[cur][wrow * 32 + s * 16 + l15][hi * 8];
            af[s].u[1] = *(const u32x4*)&As[cur][wrow * 32 + s * 16 + l15][hi * 8 + 16];
        }
        #pragma unroll
        for (int t = 0; t < 2; ++t) {
            bf[t].u[0] = *(const u32x4*)&Ws[cur][wcol * 32 + t * 16 + l15][hi * 8];
            bf[t].u[1] = *(const u32x4*)&Ws[cur][wcol * 32 + t * 16 + l15][hi * 8 + 16];
        }
        #pragma unroll
        for (int s = 0; s < 2; ++s)
            #pragma unroll
            for (int t = 0; t < 2; ++t)
                acc[s][t] = wmma_bf16(af[s].v, bf[t].v, acc[s][t]);

        if (have_next) {
            const int nxt = cur ^ 1;
            *(u32x4*)&As[nxt][ar][ac]       = a_reg;
            *(u32x4*)&Ws[nxt][wr_][wc_]     = w_reg0;
            *(u32x4*)&Ws[nxt][wr_][wc_ + 8] = w_reg1;
        }
        __syncthreads();
    }

    // epilogue: C layout -> m = g + (lane>>4)*8, n = lane&15 (+ tile offsets)
    #pragma unroll
    for (int s = 0; s < 2; ++s) {
        #pragma unroll
        for (int t = 0; t < 2; ++t) {
            const int n  = nb + wcol * 32 + t * 16 + l15;
            const float bv = bias[n];
            #pragma unroll
            for (int g = 0; g < 8; ++g) {
                const int m   = mb + wrow * 32 + s * 16 + g + hi * 8;
                const float val = acc[s][t][g] + bv;
                if (mode == 0) {
                    const int b    = m >> 11;          // m / S_  (S_ = 2048)
                    const int ss   = m & (S_ - 1);
                    const int head = n >> 6;           // n / HD_
                    const int hd   = n & (HD_ - 1);
                    outT[(((size_t)b * nh + head) * S_ + ss) * HD_ + hd] = (bf16_t)val;
                } else {
                    outF[(size_t)m * N + n] = val;
                }
            }
        }
    }
}

// ---------------------------------------------------------------------------
// Causal GQA flash attention, all matmuls via WMMA.
// block = 128 threads = 4 waves; each wave owns a 16-row query tile;
// block covers 64 query rows of one (b, h). Online softmax over 32-key tiles.
// All waves run the block-max trip count so barriers are uniform.
// ---------------------------------------------------------------------------
__global__ __launch_bounds__(128)
void flash_attn(const bf16_t* __restrict__ q, const bf16_t* __restrict__ k,
                const bf16_t* __restrict__ v, bf16_t* __restrict__ o)
{
    __shared__ bf16_t Ks[32][64];        // key-major   (key, hd)
    __shared__ bf16_t Vst[64][40];       // transposed  (hd, key), padded stride
    __shared__ bf16_t Ps[4][16][32];     // per-wave P relayout scratch

    const int tid = threadIdx.x;
    const int w   = tid >> 5;
    const int L   = tid & 31;
    const int l15 = L & 15;
    const int hi  = L >> 4;
    const int qb  = blockIdx.x * 64;
    const int h   = blockIdx.y;
    const int b   = blockIdx.z;
    const int kvh = h / G_;
    const int qt  = qb + w * 16;         // this wave's first query row

    // Q fragments (two 16x32 A-fragments covering hd 0..63), kept in VGPRs
    Frag16 qa[2];
    {
        const bf16_t* qrow = q + (((size_t)(b * H_ + h) * S_) + qt + l15) * HD_;
        #pragma unroll
        for (int kc = 0; kc < 2; ++kc) {
            qa[kc].u[0] = *(const u32x4*)(qrow + kc * 32 + hi * 8);
            qa[kc].u[1] = *(const u32x4*)(qrow + kc * 32 + hi * 8 + 16);
        }
    }

    const v8f vzero = {0,0,0,0,0,0,0,0};
    float mprev[8], lsum[8];
    v8f accO[4];
    #pragma unroll
    for (int g = 0; g < 8; ++g) { mprev[g] = -1e30f; lsum[g] = 0.f; }
    #pragma unroll
    for (int t3 = 0; t3 < 4; ++t3) accO[t3] = vzero;

    const size_t kvbase = (size_t)(b * HKV_ + kvh) * S_ * HD_;
    const int nj = qb / 32 + 2;          // keys 0 .. qb+63 (causal block max)

    for (int j = 0; j < nj; ++j) {
        // ---- cooperative K tile load (32 keys x 64 hd) ----
        {
            const int r  = tid >> 2;
            const int cb = (tid & 3) * 16;
            const bf16_t* src = k + kvbase + (size_t)(j * 32 + r) * HD_ + cb;
            *(u32x4*)&Ks[r][cb]     = *(const u32x4*)src;
            *(u32x4*)&Ks[r][cb + 8] = *(const u32x4*)(src + 8);
        }
        // ---- cooperative V tile load, transposed to (hd, key) ----
        {
            const int r  = tid & 31;
            const int hb = (tid >> 5) * 16;
            const bf16_t* src = v + kvbase + (size_t)(j * 32 + r) * HD_ + hb;
            Frag16 t;
            t.u[0] = *(const u32x4*)src;
            t.u[1] = *(const u32x4*)(src + 8);
            #pragma unroll
            for (int i = 0; i < 16; ++i) Vst[hb + i][r] = t.v[i];
        }
        // gfx1250 prefetch of next KV tile while this one is consumed
        if (j + 1 < nj) {
            const size_t nxt = kvbase + (size_t)((j + 1) * 32 + (tid & 31)) * HD_ + (tid >> 5) * 16;
            __builtin_prefetch(k + nxt, 0, 1);
            __builtin_prefetch(v + nxt, 0, 1);
        }
        __syncthreads();

        // ---- scores: Q(16x64) @ K^T(64x32) -> two 16x16 C tiles ----
        v8f sacc[2];
        #pragma unroll
        for (int t2 = 0; t2 < 2; ++t2) {
            v8f c = vzero;
            #pragma unroll
            for (int kc = 0; kc < 2; ++kc) {
                Frag16 bf;
                bf.u[0] = *(const u32x4*)&Ks[t2 * 16 + l15][kc * 32 + hi * 8];
                bf.u[1] = *(const u32x4*)&Ks[t2 * 16 + l15][kc * 32 + hi * 8 + 16];
                c = wmma_bf16(qa[kc].v, bf.v, c);
            }
            sacc[t2] = c;
        }

        // ---- scale + causal mask + online softmax ----
        float p[2][8];
        #pragma unroll
        for (int g = 0; g < 8; ++g) {
            const int qr = qt + g + hi * 8;
            #pragma unroll
            for (int t2 = 0; t2 < 2; ++t2) {
                const int key = j * 32 + t2 * 16 + l15;
                const float sv = sacc[t2][g] * 0.125f;   // 1/sqrt(64)
                p[t2][g] = (key > qr) ? -1e30f : sv;
            }
        }
        #pragma unroll
        for (int g = 0; g < 8; ++g) {
            float rm = fmaxf(p[0][g], p[1][g]);
            rm = fmaxf(rm, __shfl_xor(rm, 1));
            rm = fmaxf(rm, __shfl_xor(rm, 2));
            rm = fmaxf(rm, __shfl_xor(rm, 4));
            rm = fmaxf(rm, __shfl_xor(rm, 8));   // row max within 16-lane half
            const float mnew = fmaxf(mprev[g], rm);
            const float corr = __expf(mprev[g] - mnew);
            float rs = 0.f;
            #pragma unroll
            for (int t2 = 0; t2 < 2; ++t2) {
                const float e = __expf(p[t2][g] - mnew);
                p[t2][g] = e;
                rs += e;
            }
            rs += __shfl_xor(rs, 1);
            rs += __shfl_xor(rs, 2);
            rs += __shfl_xor(rs, 4);
            rs += __shfl_xor(rs, 8);
            lsum[g]  = lsum[g] * corr + rs;
            mprev[g] = mnew;
            #pragma unroll
            for (int t3 = 0; t3 < 4; ++t3) accO[t3][g] *= corr;
        }

        // ---- P: C-fragment layout -> A-fragment layout via LDS ----
        #pragma unroll
        for (int t2 = 0; t2 < 2; ++t2)
            #pragma unroll
            for (int g = 0; g < 8; ++g)
                Ps[w][g + hi * 8][t2 * 16 + l15] = (bf16_t)p[t2][g];
        __syncthreads();

        Frag16 pa;
        pa.u[0] = *(const u32x4*)&Ps[w][l15][hi * 8];
        pa.u[1] = *(const u32x4*)&Ps[w][l15][hi * 8 + 16];

        // ---- O += P(16x32) @ V(32x64): four 16x16 C tiles ----
        #pragma unroll
        for (int t3 = 0; t3 < 4; ++t3) {
            Frag16 vf;
            vf.u[0] = *(const u32x4*)&Vst[t3 * 16 + l15][hi * 8];
            vf.u[1] = *(const u32x4*)&Vst[t3 * 16 + l15][hi * 8 + 16];
            accO[t3] = wmma_bf16(pa.v, vf.v, accO[t3]);
        }
        __syncthreads();   // protect Ks/Vst/Ps before next tile overwrites
    }

    // ---- normalize and write O in (b, s, h, hd) layout (= O-proj A matrix) ----
    #pragma unroll
    for (int g = 0; g < 8; ++g) {
        const float inv = 1.0f / lsum[g];
        const int s = qt + g + hi * 8;
        #pragma unroll
        for (int t3 = 0; t3 < 4; ++t3) {
            const float ov = accO[t3][g] * inv;
            o[(((size_t)b * S_ + s) * H_ + h) * HD_ + t3 * 16 + l15] = (bf16_t)ov;
        }
    }
}

// ---------------------------------------------------------------------------
// host-side orchestration
// ---------------------------------------------------------------------------
extern "C" void kernel_launch(void* const* d_in, const int* in_sizes, int n_in,
                              void* d_out, int out_size, void* d_ws, size_t ws_size,
                              hipStream_t stream) {
    const float* x  = (const float*)d_in[0];
    const float* wq = (const float*)d_in[1];
    const float* bq = (const float*)d_in[2];
    const float* wk = (const float*)d_in[3];
    const float* bk = (const float*)d_in[4];
    const float* wv = (const float*)d_in[5];
    const float* bv = (const float*)d_in[6];
    const float* wo = (const float*)d_in[7];
    const float* bo = (const float*)d_in[8];
    float* out = (float*)d_out;

    char* ws = (char*)d_ws;
    const size_t MB = 1024ull * 1024ull;
    bf16_t* xb  = (bf16_t*)(ws);             // 16 MiB  (B*S x D)        bf16
    bf16_t* qb  = (bf16_t*)(ws + 16 * MB);   // 16 MiB  (B,H,S,HD)       bf16
    bf16_t* kb  = (bf16_t*)(ws + 32 * MB);   //  4 MiB  (B,HKV,S,HD)     bf16
    bf16_t* vb  = (bf16_t*)(ws + 36 * MB);   //  4 MiB  (B,HKV,S,HD)     bf16
    bf16_t* ob  = (bf16_t*)(ws + 40 * MB);   // 16 MiB  (B,S,H,HD)       bf16
    bf16_t* wqb = (bf16_t*)(ws + 56 * MB);   //  8 MiB
    bf16_t* wkb = (bf16_t*)(ws + 64 * MB);   //  2 MiB
    bf16_t* wvb = (bf16_t*)(ws + 66 * MB);   //  2 MiB
    bf16_t* wob = (bf16_t*)(ws + 68 * MB);   //  8 MiB
    (void)in_sizes; (void)n_in; (void)out_size; (void)ws_size;

    const int M = B_ * S_;   // 4096

    // one-time fp32 -> bf16 conversions (x + all weights)
    {
        int n4;
        n4 = B_ * S_ * D_ / 4;
        cvt_f32_bf16<<<(n4 + 255) / 256, 256, 0, stream>>>(x, xb, n4);
        n4 = H_ * HD_ * D_ / 4;
        cvt_f32_bf16<<<(n4 + 255) / 256, 256, 0, stream>>>(wq, wqb, n4);
        n4 = HKV_ * HD_ * D_ / 4;
        cvt_f32_bf16<<<(n4 + 255) / 256, 256, 0, stream>>>(wk, wkb, n4);
        cvt_f32_bf16<<<(n4 + 255) / 256, 256, 0, stream>>>(wv, wvb, n4);
        n4 = D_ * H_ * HD_ / 4;
        cvt_f32_bf16<<<(n4 + 255) / 256, 256, 0, stream>>>(wo, wob, n4);
    }
    // Q/K/V projections (bf16 out, head-major)
    gemm_wmma<<<dim3(H_ * HD_ / 128,   M / 64), 256, 0, stream>>>(
        xb, wqb, bq, qb, nullptr, M, H_ * HD_,   D_, 0, H_);
    gemm_wmma<<<dim3(HKV_ * HD_ / 128, M / 64), 256, 0, stream>>>(
        xb, wkb, bk, kb, nullptr, M, HKV_ * HD_, D_, 0, HKV_);
    gemm_wmma<<<dim3(HKV_ * HD_ / 128, M / 64), 256, 0, stream>>>(
        xb, wvb, bv, vb, nullptr, M, HKV_ * HD_, D_, 0, HKV_);
    // causal GQA attention
    flash_attn<<<dim3(S_ / 64, H_, B_), 128, 0, stream>>>(qb, kb, vb, ob);
    // output projection (fp32 out -> d_out)
    gemm_wmma<<<dim3(D_ / 128, M / 64), 256, 0, stream>>>(
        ob, wob, bo, nullptr, out, M, D_, H_ * HD_, 1, 0);
}
